// MetaLayer_20023137533953
// MI455X (gfx1250) — matrix operand order repaired and checked by reading, not compiled, not run on hardware
//
#include <hip/hip_runtime.h>
#include <math.h>

typedef __attribute__((ext_vector_type(16))) _Float16 v16h;
typedef __attribute__((ext_vector_type(8)))  _Float16 v8h;
typedef __attribute__((ext_vector_type(8)))  float    v8f;

constexpr int kN  = 50000;
constexpr int kNE = 400000;
constexpr int kG  = 50;
constexpr int kE  = 128;

// ---------- helpers ----------

__device__ inline void atomic_max_f32(float* addr, float val) {
  // ordered-int trick; requires init to -inf (0xFF800000)
  if (val >= 0.0f) atomicMax((int*)addr, __float_as_int(val));
  else             atomicMin((unsigned int*)addr, __float_as_uint(val));
}

// One 16x16 output tile per wave32. As: 16 x K f16 (LDS, row-major).
// Wt: 128 x K f16 (global, column-major weights = transposed W).
template<int K>
__device__ inline void wmma_tile(const _Float16* As, const _Float16* Wt,
                                 int lane, int c0, v8f& acc) {
  const int m    = lane & 15;
  const int half = lane >> 4;
  const _Float16* arow = As + m * K;
  const _Float16* bcol = Wt + (size_t)(c0 + m) * K;
#pragma unroll
  for (int k0 = 0; k0 < K; k0 += 32) {
    // A 16x32 f16 layout: lane<16 holds K runs [k0,k0+8),[k0+16,k0+24);
    // lane>=16 holds [k0+8,k0+16),[k0+24,k0+32)
    v8h a0 = *(const v8h*)(arow + k0 + half * 8);
    v8h a1 = *(const v8h*)(arow + k0 + 16 + half * 8);
    // B 32x16 f16 layout: lane<16 -> N=lane, K runs [k0,k0+8),[k0+16,k0+24); lane>=16 -> [k0+8,k0+16),[k0+24,k0+32)
    v8h b0 = *(const v8h*)(bcol + k0 + half * 8);
    v8h b1 = *(const v8h*)(bcol + k0 + 16 + half * 8);
    v16h a = __builtin_shufflevector(a0, a1, 0,1,2,3,4,5,6,7,8,9,10,11,12,13,14,15);
    v16h b = __builtin_shufflevector(b0, b1, 0,1,2,3,4,5,6,7,8,9,10,11,12,13,14,15);
    acc = __builtin_amdgcn_wmma_f32_16x16x32_f16(false, a, false, b, (short)0, acc,
                                                 false, false);
  }
}

// ---------- utility kernels ----------

__global__ void k_fill(float* p, long long n, float v) {
  long long i = (long long)blockIdx.x * 256 + threadIdx.x;
  if (i < n) p[i] = v;
}

// Wt[c*K + k] = (f16) W[k*128 + c]   (W is K x 128 row-major)
__global__ void k_w_to_f16t(const float* W, _Float16* Wt, int K) {
  int i = blockIdx.x * 256 + threadIdx.x;
  if (i < K * 128) {
    int k = i >> 7, c = i & 127;
    Wt[(size_t)c * K + k] = (_Float16)W[i];
  }
}

// ---------- GEMM kernels ----------

// e_new = [edge_attr | x[src] | x[dst] | u[ebatch]] @ edge_W + edge_b
__global__ __launch_bounds__(256)
void k_gemm_edge(const float* edge_attr, const float* x, const float* u,
                 const int* src, const int* dst, const int* ebatch,
                 const _Float16* Wt, const float* bias, float* out) {
  __shared__ _Float16 As[16 * 512];
  const int rb = blockIdx.x * 16, tid = threadIdx.x;
  for (int i = tid; i < 16 * 512; i += 256) {
    int r = i >> 9, c = i & 511;
    int e = rb + r; float v = 0.0f;
    if (e < kNE) {
      if      (c < 128) v = edge_attr[(size_t)e * 128 + c];
      else if (c < 256) v = x[(size_t)src[e] * 128 + (c - 128)];
      else if (c < 384) v = x[(size_t)dst[e] * 128 + (c - 256)];
      else              v = u[(size_t)ebatch[e] * 128 + (c - 384)];
    }
    As[i] = (_Float16)v;
  }
  __syncthreads();
  const int lane = tid & 31, c0 = (tid >> 5) * 16;
  const int m = lane & 15, half = lane >> 4;
  v8f acc = {};
  wmma_tile<512>(As, Wt, lane, c0, acc);
  const float bv = bias[c0 + m];
#pragma unroll
  for (int i = 0; i < 8; ++i) {
    int e = rb + i + half * 8;
    if (e < kNE) out[(size_t)e * 128 + c0 + m] = acc[i] + bv;
  }
}

// node-attention pass A: h = lrelu([x[qi]|x[ki]|e_new] @ na_w1 + b1),
// logits = einsum(h, w2); store logits, atomic-max per segment.
__global__ __launch_bounds__(256)
void k_nattn_a(const float* x, const float* enew,
               const int* qi, const int* ki, const int* idx,
               const _Float16* Wt, const float* bias, const float* w2,
               float* logits, float* segmax) {
  __shared__ _Float16 As[16 * 384];
  __shared__ float sl[32];
  const int rb = blockIdx.x * 16, tid = threadIdx.x;
  for (int i = tid; i < 16 * 384; i += 256) {
    int r = i / 384, c = i % 384;
    int e = rb + r; float v = 0.0f;
    if (e < kNE) {
      if      (c < 128) v = x[(size_t)qi[e] * 128 + c];
      else if (c < 256) v = x[(size_t)ki[e] * 128 + (c - 128)];
      else              v = enew[(size_t)e * 128 + (c - 256)];
    }
    As[i] = (_Float16)v;
  }
  if (tid < 32) sl[tid] = 0.0f;
  __syncthreads();
  const int lane = tid & 31, c0 = (tid >> 5) * 16;
  const int m = lane & 15, half = lane >> 4;
  v8f acc = {};
  wmma_tile<384>(As, Wt, lane, c0, acc);
  const float bv = bias[c0 + m];
  const float wv = w2[c0 + m];      // w2 flat (H*D = 128), col -> head*64+d
  const int head = c0 >> 6;
#pragma unroll
  for (int i = 0; i < 8; ++i) {
    float hraw = acc[i] + bv;
    float hval = hraw > 0.0f ? hraw : 0.01f * hraw;  // leaky_relu(0.01)
    float p = hval * wv;
    p += __shfl_xor(p, 1, 32);
    p += __shfl_xor(p, 2, 32);
    p += __shfl_xor(p, 4, 32);
    p += __shfl_xor(p, 8, 32);
    if (m == 0) atomicAdd(&sl[(i + half * 8) * 2 + head], p);
  }
  __syncthreads();
  if (tid < 32) {
    int r = tid >> 1, h = tid & 1;
    int e = rb + r;
    if (e < kNE) {
      float l = sl[tid];
      logits[(size_t)e * 2 + h] = l;
      atomic_max_f32(&segmax[(size_t)idx[e] * 2 + h], l);
    }
  }
}

// softmax pass B (generic): ex = exp(l - max[seg]); sum[seg] += ex
__global__ void k_seg_exp(float* logits, const int* idx, const float* segmax,
                          float* segsum, long long n) {
  long long i = (long long)blockIdx.x * 256 + threadIdx.x;
  if (i < n * 2) {
    long long e = i >> 1; int h = (int)(i & 1);
    int t = idx[e];
    float ex = expf(logits[i] - segmax[(size_t)t * 2 + h]);
    logits[i] = ex;
    atomicAdd(&segsum[(size_t)t * 2 + h], ex);
  }
}

// node-attention pass C: v = [x[ki]|e_new] @ na_w3 + b3, scatter attn*v into out[idx]
__global__ __launch_bounds__(256)
void k_nattn_c(const float* x, const float* enew,
               const int* ki, const int* idx,
               const _Float16* Wt, const float* bias,
               const float* ex, const float* segsum, float* out) {
  __shared__ _Float16 As[16 * 256];
  const int rb = blockIdx.x * 16, tid = threadIdx.x;
  for (int i = tid; i < 16 * 256; i += 256) {
    int r = i >> 8, c = i & 255;
    int e = rb + r; float v = 0.0f;
    if (e < kNE) {
      if (c < 128) v = x[(size_t)ki[e] * 128 + c];
      else         v = enew[(size_t)e * 128 + (c - 128)];
    }
    As[i] = (_Float16)v;
  }
  __syncthreads();
  const int lane = tid & 31, c0 = (tid >> 5) * 16;
  const int m = lane & 15, half = lane >> 4;
  v8f acc = {};
  wmma_tile<256>(As, Wt, lane, c0, acc);
  const int col = c0 + m, head = col >> 6;
  const float bv = bias[col];
#pragma unroll
  for (int i = 0; i < 8; ++i) {
    int e = rb + i + half * 8;
    if (e < kNE) {
      int t = idx[e];
      float a = ex[(size_t)e * 2 + head] / (segsum[(size_t)t * 2 + head] + 1e-16f);
      atomicAdd(&out[(size_t)t * 128 + col], (acc[i] + bv) * a);
    }
  }
}

// x_new = [x | sent | recv | u[nbatch]] @ node_W + node_b
__global__ __launch_bounds__(256)
void k_gemm_node(const float* x, const float* sent, const float* recv,
                 const float* u, const int* nbatch,
                 const _Float16* Wt, const float* bias, float* out) {
  __shared__ _Float16 As[16 * 512];
  const int rb = blockIdx.x * 16, tid = threadIdx.x;
  for (int i = tid; i < 16 * 512; i += 256) {
    int r = i >> 9, c = i & 511;
    int n = rb + r; float v = 0.0f;
    if (n < kN) {
      if      (c < 128) v = x[(size_t)n * 128 + c];
      else if (c < 256) v = sent[(size_t)n * 128 + (c - 128)];
      else if (c < 384) v = recv[(size_t)n * 128 + (c - 256)];
      else              v = u[(size_t)nbatch[n] * 128 + (c - 384)];
    }
    As[i] = (_Float16)v;
  }
  __syncthreads();
  const int lane = tid & 31, c0 = (tid >> 5) * 16;
  const int m = lane & 15, half = lane >> 4;
  v8f acc = {};
  wmma_tile<512>(As, Wt, lane, c0, acc);
  const float bv = bias[c0 + m];
#pragma unroll
  for (int i = 0; i < 8; ++i) {
    int n = rb + i + half * 8;
    if (n < kN) out[(size_t)n * 128 + c0 + m] = acc[i] + bv;
  }
}

// global-attention pass A: h = lrelu([u[batch]|feat] @ w1 + b1); logits; seg max
__global__ __launch_bounds__(256)
void k_gattn_a(const float* u, const float* feat, const int* batch,
               const _Float16* Wt, const float* bias, const float* w2,
               float* logits, float* gmax, int nrows) {
  __shared__ _Float16 As[16 * 256];
  __shared__ float sl[32];
  const int rb = blockIdx.x * 16, tid = threadIdx.x;
  for (int i = tid; i < 16 * 256; i += 256) {
    int r = i >> 8, c = i & 255;
    int n = rb + r; float v = 0.0f;
    if (n < nrows) {
      if (c < 128) v = u[(size_t)batch[n] * 128 + c];
      else         v = feat[(size_t)n * 128 + (c - 128)];
    }
    As[i] = (_Float16)v;
  }
  if (tid < 32) sl[tid] = 0.0f;
  __syncthreads();
  const int lane = tid & 31, c0 = (tid >> 5) * 16;
  const int m = lane & 15, half = lane >> 4;
  v8f acc = {};
  wmma_tile<256>(As, Wt, lane, c0, acc);
  const float bv = bias[c0 + m];
  const float wv = w2[c0 + m];
  const int head = c0 >> 6;
#pragma unroll
  for (int i = 0; i < 8; ++i) {
    float hraw = acc[i] + bv;
    float hval = hraw > 0.0f ? hraw : 0.01f * hraw;
    float p = hval * wv;
    p += __shfl_xor(p, 1, 32);
    p += __shfl_xor(p, 2, 32);
    p += __shfl_xor(p, 4, 32);
    p += __shfl_xor(p, 8, 32);
    if (m == 0) atomicAdd(&sl[(i + half * 8) * 2 + head], p);
  }
  __syncthreads();
  if (tid < 32) {
    int r = tid >> 1, h = tid & 1;
    int n = rb + r;
    if (n < nrows) {
      float l = sl[tid];
      logits[(size_t)n * 2 + h] = l;
      atomic_max_f32(&gmax[(size_t)batch[n] * 2 + h], l);
    }
  }
}

// global-attention pass C: out[g] += attn * feat (v == feat, no GEMM)
__global__ void k_gattn_c(const float* feat, const int* batch,
                          const float* ex, const float* gsum,
                          float* out, int nrows) {
  const int tid = threadIdx.x;
  const long long r0 = (long long)blockIdx.x * 64;
  const int c = tid & 127;
  const int rr = tid >> 7;
  const int head = c >> 6;
  const int b0 = batch[r0];
  float local = 0.0f;
  for (int i = 0; i < 32; ++i) {
    long long r = r0 + rr + 2 * i;
    if (r < nrows) {
      int b = batch[r];
      float w = ex[(size_t)r * 2 + head] / (gsum[(size_t)b * 2 + head] + 1e-16f);
      float v = feat[(size_t)r * 128 + c] * w;
      if (b == b0) local += v;
      else atomicAdd(&out[(size_t)b * 128 + c], v);
    }
  }
  atomicAdd(&out[(size_t)b0 * 128 + c], local);
}

// u_new = [u | na | ea] @ glob_W + glob_b
__global__ __launch_bounds__(256)
void k_gemm_glob(const float* u, const float* na, const float* ea,
                 const _Float16* Wt, const float* bias, float* out) {
  __shared__ _Float16 As[16 * 384];
  const int rb = blockIdx.x * 16, tid = threadIdx.x;
  for (int i = tid; i < 16 * 384; i += 256) {
    int r = i / 384, c = i % 384;
    int g = rb + r; float v = 0.0f;
    if (g < kG) {
      if      (c < 128) v = u[(size_t)g * 128 + c];
      else if (c < 256) v = na[(size_t)g * 128 + (c - 128)];
      else              v = ea[(size_t)g * 128 + (c - 256)];
    }
    As[i] = (_Float16)v;
  }
  __syncthreads();
  const int lane = tid & 31, c0 = (tid >> 5) * 16;
  const int m = lane & 15, half = lane >> 4;
  v8f acc = {};
  wmma_tile<384>(As, Wt, lane, c0, acc);
  const float bv = bias[c0 + m];
#pragma unroll
  for (int i = 0; i < 8; ++i) {
    int g = rb + i + half * 8;
    if (g < kG) out[(size_t)g * 128 + c0 + m] = acc[i] + bv;
  }
}

// ---------- host ----------

extern "C" void kernel_launch(void* const* d_in, const int* in_sizes, int n_in,
                              void* d_out, int out_size, void* d_ws, size_t ws_size,
                              hipStream_t stream) {
  (void)in_sizes; (void)n_in; (void)out_size; (void)ws_size;
  const float* x         = (const float*)d_in[0];
  const float* edge_attr = (const float*)d_in[1];
  const float* u         = (const float*)d_in[2];
  const float* edge_W    = (const float*)d_in[3];
  const float* edge_b    = (const float*)d_in[4];
  const float* node_W    = (const float*)d_in[5];
  const float* node_b    = (const float*)d_in[6];
  const float* glob_W    = (const float*)d_in[7];
  const float* glob_b    = (const float*)d_in[8];
  const float* na_w1     = (const float*)d_in[9];
  const float* na_b1     = (const float*)d_in[10];
  const float* na_w2     = (const float*)d_in[11];
  const float* na_w3     = (const float*)d_in[12];
  const float* na_b3     = (const float*)d_in[13];
  const float* gn_w1     = (const float*)d_in[14];
  const float* gn_b1     = (const float*)d_in[15];
  const float* gn_w2     = (const float*)d_in[16];
  const float* ge_w1     = (const float*)d_in[17];
  const float* ge_b1     = (const float*)d_in[18];
  const float* ge_w2     = (const float*)d_in[19];
  const int*   eidx      = (const int*)d_in[20];
  const int*   nbatch    = (const int*)d_in[21];
  const int*   ebatch    = (const int*)d_in[22];
  const int*   src = eidx;          // row = edge_index[0]
  const int*   dst = eidx + kNE;    // col = edge_index[1]

  float* x_new = (float*)d_out;
  float* e_new = x_new + (size_t)kN * kE;
  float* u_new = e_new + (size_t)kNE * kE;

  // workspace carve-up (256B aligned)
  char* ws = (char*)d_ws;
  size_t cur = 0;
  auto take = [&](size_t bytes) -> char* {
    char* p = ws + cur;
    cur = (cur + bytes + 255) & ~(size_t)255;
    return p;
  };
  _Float16* wt_edge = (_Float16*)take((size_t)512 * 128 * 2);
  _Float16* wt_node = (_Float16*)take((size_t)512 * 128 * 2);
  _Float16* wt_glob = (_Float16*)take((size_t)384 * 128 * 2);
  _Float16* wt_na1  = (_Float16*)take((size_t)384 * 128 * 2);
  _Float16* wt_na3  = (_Float16*)take((size_t)256 * 128 * 2);
  _Float16* wt_gn1  = (_Float16*)take((size_t)256 * 128 * 2);
  _Float16* wt_ge1  = (_Float16*)take((size_t)256 * 128 * 2);
  float* sent   = (float*)take((size_t)kN * kE * 4);
  float* recv   = (float*)take((size_t)kN * kE * 4);
  float* smax_s = (float*)take((size_t)kN * 2 * 4);
  float* ssum_s = (float*)take((size_t)kN * 2 * 4);
  float* smax_r = (float*)take((size_t)kN * 2 * 4);
  float* ssum_r = (float*)take((size_t)kN * 2 * 4);
  float* elog_s = (float*)take((size_t)kNE * 2 * 4);
  float* elog_r = (float*)take((size_t)kNE * 2 * 4);
  float* gmax_n = (float*)take((size_t)kG * 2 * 4);
  float* gsum_n = (float*)take((size_t)kG * 2 * 4);
  float* gmax_e = (float*)take((size_t)kG * 2 * 4);
  float* gsum_e = (float*)take((size_t)kG * 2 * 4);
  float* glog_n = (float*)take((size_t)kN * 2 * 4);
  float* glog_e = (float*)take((size_t)kNE * 2 * 4);
  float* na     = (float*)take((size_t)kG * kE * 4);
  float* ea     = (float*)take((size_t)kG * kE * 4);

  auto cdiv = [](long long a, long long b) { return (unsigned)((a + b - 1) / b); };
  auto fill = [&](float* p, long long n, float v) {
    k_fill<<<cdiv(n, 256), 256, 0, stream>>>(p, n, v);
  };

  // weight conversion (f32 row-major -> f16 col-major)
  k_w_to_f16t<<<cdiv(512 * 128, 256), 256, 0, stream>>>(edge_W, wt_edge, 512);
  k_w_to_f16t<<<cdiv(512 * 128, 256), 256, 0, stream>>>(node_W, wt_node, 512);
  k_w_to_f16t<<<cdiv(384 * 128, 256), 256, 0, stream>>>(glob_W, wt_glob, 384);
  k_w_to_f16t<<<cdiv(384 * 128, 256), 256, 0, stream>>>(na_w1,  wt_na1,  384);
  k_w_to_f16t<<<cdiv(256 * 128, 256), 256, 0, stream>>>(na_w3,  wt_na3,  256);
  k_w_to_f16t<<<cdiv(256 * 128, 256), 256, 0, stream>>>(gn_w1,  wt_gn1,  256);
  k_w_to_f16t<<<cdiv(256 * 128, 256), 256, 0, stream>>>(ge_w1,  wt_ge1,  256);

  // re-init accumulation buffers every launch (deterministic)
  fill(sent,   (long long)kN * kE, 0.0f);
  fill(recv,   (long long)kN * kE, 0.0f);
  fill(ssum_s, (long long)kN * 2, 0.0f);
  fill(ssum_r, (long long)kN * 2, 0.0f);
  fill(smax_s, (long long)kN * 2, -HUGE_VALF);
  fill(smax_r, (long long)kN * 2, -HUGE_VALF);
  fill(gsum_n, (long long)kG * 2, 0.0f);
  fill(gsum_e, (long long)kG * 2, 0.0f);
  fill(gmax_n, (long long)kG * 2, -HUGE_VALF);
  fill(gmax_e, (long long)kG * 2, -HUGE_VALF);
  fill(na,     (long long)kG * kE, 0.0f);
  fill(ea,     (long long)kG * kE, 0.0f);

  const unsigned gE = cdiv(kNE, 16), gNn = cdiv(kN, 16), gGg = cdiv(kG, 16);

  // 1) e_new
  k_gemm_edge<<<gE, 256, 0, stream>>>(edge_attr, x, u, src, dst, ebatch,
                                      wt_edge, edge_b, e_new);
  // 2) node attention (sent: q=x[src], kv=x[dst], idx=src)
  k_nattn_a<<<gE, 256, 0, stream>>>(x, e_new, src, dst, src,
                                    wt_na1, na_b1, na_w2, elog_s, smax_s);
  // 3) node attention (recv: q=x[dst], kv=x[src], idx=dst)
  k_nattn_a<<<gE, 256, 0, stream>>>(x, e_new, dst, src, dst,
                                    wt_na1, na_b1, na_w2, elog_r, smax_r);
  k_seg_exp<<<cdiv((long long)kNE * 2, 256), 256, 0, stream>>>(elog_s, src, smax_s, ssum_s, kNE);
  k_seg_exp<<<cdiv((long long)kNE * 2, 256), 256, 0, stream>>>(elog_r, dst, smax_r, ssum_r, kNE);
  k_nattn_c<<<gE, 256, 0, stream>>>(x, e_new, dst, src, wt_na3, na_b3,
                                    elog_s, ssum_s, sent);
  k_nattn_c<<<gE, 256, 0, stream>>>(x, e_new, src, dst, wt_na3, na_b3,
                                    elog_r, ssum_r, recv);
  // 4) x_new
  k_gemm_node<<<gNn, 256, 0, stream>>>(x, sent, recv, u, nbatch,
                                       wt_node, node_b, x_new);
  // 5) global attention over nodes (na) and edges (ea)
  k_gattn_a<<<gNn, 256, 0, stream>>>(u, x_new, nbatch, wt_gn1, gn_b1, gn_w2,
                                     glog_n, gmax_n, kN);
  k_gattn_a<<<gE, 256, 0, stream>>>(u, e_new, ebatch, wt_ge1, ge_b1, ge_w2,
                                    glog_e, gmax_e, kNE);
  k_seg_exp<<<cdiv((long long)kN * 2, 256), 256, 0, stream>>>(glog_n, nbatch, gmax_n, gsum_n, kN);
  k_seg_exp<<<cdiv((long long)kNE * 2, 256), 256, 0, stream>>>(glog_e, ebatch, gmax_e, gsum_e, kNE);
  k_gattn_c<<<cdiv(kN, 64), 256, 0, stream>>>(x_new, nbatch, glog_n, gsum_n, na, kN);
  k_gattn_c<<<cdiv(kNE, 64), 256, 0, stream>>>(e_new, ebatch, glog_e, gsum_e, ea, kNE);
  // 6) u_new
  k_gemm_glob<<<gGg, 256, 0, stream>>>(u, na, ea, wt_glob, glob_b, u_new);
}
